// Som_28784870817792
// MI455X (gfx1250) — compile-verified, matrix operand order checked by reading, not run on hardware
//
#include <hip/hip_runtime.h>
#include <hip/hip_bf16.h>

// CDNA5 / gfx1250 SOM training step.
//   GEMM1 (BMU):    S = X (256x128) * W^T (128x4096)   -> v_wmma_f32_16x16x32_bf16
//   GEMM2 (update): U = N^T (4096x256) * X (256x128)   -> v_wmma_f32_16x16x32_bf16
//   new_w[n,d] = w[n,d] + (alpha/B) * (U[n,d] - s[n]*w[n,d]),  s[n] = sum_b N[b,n]

typedef __attribute__((ext_vector_type(16))) __bf16 v16bf;
typedef __attribute__((ext_vector_type(8)))  float  v8f;

#define R_DIM 64
#define C_DIM 64
#define D_DIM 128
#define B_DIM 256
#define RC    (R_DIM * C_DIM)     // 4096
#define NT    (RC / 16)           // 256 neuron tiles
#define KC1   (D_DIM / 32)        // 4  K-chunks for GEMM1 (K = D)
#define KC2   (B_DIM / 32)        // 8  K-chunks for GEMM2 (K = B)
#define DT    (D_DIM / 16)        // 8  d tiles
#define BT    (B_DIM / 16)        // 16 batch tiles
#define NSPLIT 16                 // n-range splits for the BMU kernel
#define TPS   (NT / NSPLIT / 8)   // 2 tiles per wave per split
#define FRAG_ELEMS 512            // 32 lanes * 16 bf16 per fragment

__device__ __forceinline__ __bf16 f32_to_bf16(float f) {
  unsigned u = __builtin_bit_cast(unsigned, f);
  unsigned r = (u + 0x7FFFu + ((u >> 16) & 1u)) >> 16;   // RNE
  return __builtin_bit_cast(__bf16, (unsigned short)r);
}
__device__ __forceinline__ float bf16_to_f32(__bf16 h) {
  unsigned u = ((unsigned)__builtin_bit_cast(unsigned short, h)) << 16;
  return __builtin_bit_cast(float, u);
}
// Monotone float->u32 key: preserves total order, so u64 (key<<32 | idx)
// atomicMin == (min score, then min index) == jnp.argmin semantics.
__device__ __forceinline__ unsigned fkey(float f) {
  unsigned u = __builtin_bit_cast(unsigned, f);
  return (u & 0x80000000u) ? ~u : (u | 0x80000000u);
}

// ISA 7.12.2 16-bit A/B fragment K mapping: element e of the 16-element
// per-lane vector; low half-wave holds K {0..7,16..23}, high half {8..15,24..31}.
__device__ __forceinline__ int kmap(int lane, int e) {
  return ((e < 8) ? e : e + 8) + ((lane & 16) ? 8 : 0);
}

// ---------------------------------------------------------------------------
// Pack W (4096x128 f32) into bf16 B-operand fragments (N=neuron, K=d) and wnorm.
// grid 256 (nt), block 128 (4 waves = 4 K-chunks).
__global__ void k_pack_w(const float* __restrict__ w,
                         __bf16* __restrict__ wpack,
                         float* __restrict__ wnorm) {
  __shared__ float tile[16 * D_DIM];
  const int nt = blockIdx.x, t = threadIdx.x;
  const float4* src = (const float4*)(w + nt * 16 * D_DIM);
  float4* dst = (float4*)tile;
#pragma unroll
  for (int i = 0; i < 4; ++i) dst[t + 128 * i] = src[t + 128 * i];
  __syncthreads();

  const int kc = t >> 5, lane = t & 31, idx = lane & 15;
  v16bf f;
#pragma unroll
  for (int e = 0; e < 16; ++e) {
    int k = kc * 32 + kmap(lane, e);
    f[e] = f32_to_bf16(tile[idx * D_DIM + k]);
  }
  *(v16bf*)(wpack + (nt * KC1 + kc) * FRAG_ELEMS + lane * 16) = f;

  if (t < 16) {
    float s = 0.f;
#pragma unroll 4
    for (int d = 0; d < D_DIM; ++d) { float v = tile[t * D_DIM + d]; s += v * v; }
    wnorm[nt * 16 + t] = s;
  }
}

// ---------------------------------------------------------------------------
// Pack X both as A-operand (M=b, K=d) for GEMM1 and as B-operand (K=b, N=d)
// for GEMM2; also initializes the 256 argmin keys (runs before k_bmu in-stream).
// grid 24, block 128: blocks 0..15 -> bt A-pack, 16..23 -> dt B-pack.
__global__ void k_pack_x(const float* __restrict__ x,
                         __bf16* __restrict__ xpackA,
                         __bf16* __restrict__ xpackB,
                         unsigned long long* __restrict__ keys) {
  __shared__ float tile[4096];
  const int t = threadIdx.x, kcw = t >> 5, lane = t & 31, idx = lane & 15;
  if (blockIdx.x < 2) keys[blockIdx.x * 128 + t] = ~0ull;
  if (blockIdx.x < BT) {
    const int bt = blockIdx.x;                 // 16 x 128 row tile of x
    const float4* src = (const float4*)(x + bt * 16 * D_DIM);
    float4* dst = (float4*)tile;
#pragma unroll
    for (int i = 0; i < 4; ++i) dst[t + 128 * i] = src[t + 128 * i];
    __syncthreads();
    v16bf f;
#pragma unroll
    for (int e = 0; e < 16; ++e) {
      int k = kcw * 32 + kmap(lane, e);
      f[e] = f32_to_bf16(tile[idx * D_DIM + k]);
    }
    *(v16bf*)(xpackA + (bt * KC1 + kcw) * FRAG_ELEMS + lane * 16) = f;
  } else {
    const int dt = blockIdx.x - BT;            // 256 x 16 column slab of x
    for (int i = t; i < 1024; i += 128) {      // 1024 float4 loads
      int b = i >> 2, q = i & 3;
      float4 v = *(const float4*)(x + b * D_DIM + dt * 16 + q * 4);
      *(float4*)(tile + b * 16 + q * 4) = v;
    }
    __syncthreads();
#pragma unroll
    for (int c = 0; c < 2; ++c) {
      int kc = kcw * 2 + c;
      v16bf f;
#pragma unroll
      for (int e = 0; e < 16; ++e) {
        int b = kc * 32 + kmap(lane, e);
        f[e] = f32_to_bf16(tile[b * 16 + idx]);
      }
      *(v16bf*)(xpackB + (dt * KC2 + kc) * FRAG_ELEMS + lane * 16) = f;
    }
  }
}

// ---------------------------------------------------------------------------
// GEMM1 + fused argmin: score(b,n) = ||w_n||^2 - 2 * x_b . w_n.
// grid (NSPLIT, BT), block 256 (8 waves). Each wave owns 2 tiles; ALL their
// B-fragments are preloaded up front (flat SSA live ranges, no rotation, no
// spills) so the single load clause drains under the 8 WMMAs via staged
// s_wait_loadcnt. Partials combine via deterministic u64 atomicMin keys.
__global__ void k_bmu(const __bf16* __restrict__ xpackA,
                      const __bf16* __restrict__ wpackB,
                      const float* __restrict__ wnorm,
                      unsigned long long* __restrict__ keys) {
  __shared__ float rv[8][16];
  __shared__ int   ri[8][16];
  const int sp = blockIdx.x, bt = blockIdx.y;
  const int t = threadIdx.x, wv = t >> 5, lane = t & 31;
  const int nloc = lane & 15;
  const int base = sp * (NT / NSPLIT);         // 16-tile slice

  v16bf a[KC1];
#pragma unroll
  for (int kc = 0; kc < KC1; ++kc)
    a[kc] = *(const v16bf*)(xpackA + (bt * KC1 + kc) * FRAG_ELEMS + lane * 16);

  // preload both tiles' fragments + wnorm: one big clause, no buffer rotation
  v16bf b[TPS][KC1];
  float wn[TPS];
#pragma unroll
  for (int k = 0; k < TPS; ++k) {
    const int nt = base + wv + 8 * k;
    const __bf16* np = wpackB + (nt * KC1) * FRAG_ELEMS + lane * 16;
#pragma unroll
    for (int kc = 0; kc < KC1; ++kc)
      b[k][kc] = *(const v16bf*)(np + kc * FRAG_ELEMS);
    wn[k] = wnorm[nt * 16 + nloc];
  }

  float mv[8]; int mi[8];
#pragma unroll
  for (int r = 0; r < 8; ++r) { mv[r] = 3.4e38f; mi[r] = 0x7fffffff; }

#pragma unroll
  for (int k = 0; k < TPS; ++k) {
    const int nt = base + wv + 8 * k;
    v8f acc = {};
#pragma unroll
    for (int kc = 0; kc < KC1; ++kc)
      acc = __builtin_amdgcn_wmma_f32_16x16x32_bf16(false, a[kc], false, b[k][kc],
                                                    (short)0, acc, false, false);
    const int ni = nt * 16 + nloc;
#pragma unroll
    for (int r = 0; r < 8; ++r) {
      float sc = fmaf(-2.0f, acc[r], wn[k]);    // strict <: lean hot loop
      if (sc < mv[r]) { mv[r] = sc; mi[r] = ni; }
    }
  }

  // reduce across the 16 lanes of each half-wave (the N dimension)
#pragma unroll
  for (int m = 1; m < 16; m <<= 1) {
#pragma unroll
    for (int r = 0; r < 8; ++r) {
      float ov = __shfl_xor(mv[r], m, 32);
      int   oi = __shfl_xor(mi[r], m, 32);
      if (ov < mv[r] || (ov == mv[r] && oi < mi[r])) { mv[r] = ov; mi[r] = oi; }
    }
  }
  if (lane == 0)
#pragma unroll
    for (int r = 0; r < 8; ++r) { rv[wv][r] = mv[r]; ri[wv][r] = mi[r]; }
  if (lane == 16)
#pragma unroll
    for (int r = 0; r < 8; ++r) { rv[wv][r + 8] = mv[r]; ri[wv][r + 8] = mi[r]; }
  __syncthreads();
  if (t < 16) {
    float bv = rv[0][t]; int bi = ri[0][t];
#pragma unroll
    for (int j = 1; j < 8; ++j) {
      float ov = rv[j][t]; int oi = ri[j][t];
      if (ov < bv || (ov == bv && oi < bi)) { bv = ov; bi = oi; }
    }
    unsigned long long key = ((unsigned long long)fkey(bv) << 32) | (unsigned)bi;
    atomicMin(&keys[bt * 16 + t], key);        // order-independent combine
  }
}

// ---------------------------------------------------------------------------
// GEMM2 + weight update. grid 256 (nt), block 256 (8 waves; wave wv = d-tile).
// Neighborhood A-fragments are built value-parallel into LDS from BMU coords.
__global__ void k_update(const float* __restrict__ w,
                         const __bf16* __restrict__ xpackB,
                         const unsigned* __restrict__ keys32,  // low dword = bmu idx
                         const float* __restrict__ alpha_p,
                         const float* __restrict__ sigma_p,
                         float* __restrict__ out) {
  __shared__ __bf16 nfrag[KC2 * FRAG_ELEMS];   // 8 KB
  __shared__ float  s_lds[8][16];
  const int nt = blockIdx.x, t = threadIdx.x, wv = t >> 5, lane = t & 31;
  const int nloc = lane & 15;

  // pull this block's 8KB weight tile toward the caches while we compute
  __builtin_prefetch(w + nt * 16 * D_DIM + t * 8, 0, 1);

  const float sg = sigma_p[0];
  const float inv_s2 = 1.0f / (sg * sg);
  const float coef = alpha_p[0] * (1.0f / (float)B_DIM);

  const int ng_lane = nt * 16 + nloc;
  const float pr = (float)(ng_lane >> 6), pc = (float)(ng_lane & 63);

  // wave wv builds K-chunk wv of N^T : A[n, b] = exp(-||bmu_b - pos_n||^2/s^2)
  v16bf f; float ssum = 0.f;
#pragma unroll
  for (int e = 0; e < 16; ++e) {
    int b = wv * 32 + kmap(lane, e);
    int bm = (int)keys32[2 * b];               // low 32 bits of the argmin key
    float dr = (float)(bm >> 6) - pr;
    float dc = (float)(bm & 63) - pc;
    float nv = expf(-(dr * dr + dc * dc) * inv_s2);
    __bf16 h = f32_to_bf16(nv);
    f[e] = h;
    ssum += bf16_to_f32(h);        // accumulate s[n] at the precision the GEMM sees
  }
  *(v16bf*)(nfrag + wv * FRAG_ELEMS + lane * 16) = f;
  ssum += __shfl_xor(ssum, 16, 32);   // fold the two K half-waves
  if (lane < 16) s_lds[wv][lane] = ssum;
  __syncthreads();

  // U tile: 16 n x 16 d, K = 256 over 8 chunks
  v8f acc = {};
#pragma unroll
  for (int kc = 0; kc < KC2; ++kc) {
    v16bf af = *(const v16bf*)(nfrag + kc * FRAG_ELEMS + lane * 16);
    v16bf bf = *(const v16bf*)(xpackB + (wv * KC2 + kc) * FRAG_ELEMS + lane * 16);
    acc = __builtin_amdgcn_wmma_f32_16x16x32_bf16(false, af, false, bf,
                                                  (short)0, acc, false, false);
  }

  const int dcol = wv * 16 + nloc;
  const int half = (lane >> 4) << 3;   // C layout: M = r + 8*(lane>=16)
#pragma unroll
  for (int r = 0; r < 8; ++r) {
    int M = r + half;
    int ng = nt * 16 + M;
    float s = 0.f;
#pragma unroll
    for (int j = 0; j < 8; ++j) s += s_lds[j][M];
    float wo = w[ng * D_DIM + dcol];
    out[ng * D_DIM + dcol] = wo + coef * (acc[r] - s * wo);
  }
}

// ---------------------------------------------------------------------------
extern "C" void kernel_launch(void* const* d_in, const int* in_sizes, int n_in,
                              void* d_out, int out_size, void* d_ws, size_t ws_size,
                              hipStream_t stream) {
  const float* x     = (const float*)d_in[0];   // [256,128]
  const float* wts   = (const float*)d_in[1];   // [64,64,128]
  const float* alpha = (const float*)d_in[2];   // scalar
  const float* sigma = (const float*)d_in[3];   // scalar
  float* out = (float*)d_out;

  char* ws = (char*)d_ws;
  float*  wnorm = (float*)(ws);                               // 16 KB
  unsigned long long* keys = (unsigned long long*)(ws + 16384); // 2 KB
  __bf16* xpackA = (__bf16*)(ws + 32768);                     // 64 KB
  __bf16* xpackB = (__bf16*)(ws + 32768 + 65536);             // 64 KB
  __bf16* wpackB = (__bf16*)(ws + 32768 + 131072);            // 1 MB

  k_pack_w <<<NT,      128, 0, stream>>>(wts, wpackB, wnorm);
  k_pack_x <<<BT + DT, 128, 0, stream>>>(x, xpackA, xpackB, keys);
  k_bmu    <<<dim3(NSPLIT, BT), 256, 0, stream>>>(xpackA, wpackB, wnorm, keys);
  k_update <<<NT,      256, 0, stream>>>(wts, xpackB, (const unsigned*)keys,
                                         alpha, sigma, out);
}
